// SemanticVQVAE_64493228916939
// MI455X (gfx1250) — matrix operand order repaired and checked by reading, not compiled, not run on hardware
//
#include <hip/hip_runtime.h>
#include <hip/hip_bf16.h>

// ---------------------------------------------------------------------------
// Types for WMMA (CDNA5 / gfx1250, wave32)
// ---------------------------------------------------------------------------
typedef __attribute__((ext_vector_type(16))) __bf16          v16bf;
typedef __attribute__((ext_vector_type(8)))  float           v8f;
typedef __attribute__((ext_vector_type(16))) unsigned short  v16us;
typedef __attribute__((ext_vector_type(8)))  unsigned short  v8us;

typedef unsigned short bf16s;   // raw bf16 storage

__device__ __forceinline__ bf16s f2bf(float f) {
    union { float f; unsigned u; } x; x.f = f;
    unsigned u = x.u;
    unsigned r = u + 0x7fffu + ((u >> 16) & 1u);   // round-to-nearest-even
    return (bf16s)(r >> 16);
}

__device__ __forceinline__ v8f wmma_bf16(v16us a, v16us b, v8f c) {
    return __builtin_amdgcn_wmma_f32_16x16x32_bf16(
        false, __builtin_bit_cast(v16bf, a),
        false, __builtin_bit_cast(v16bf, b),
        (short)0, c, false, false);
}

// Low 32 bits of a flat pointer into LDS == wave-relative LDS byte address
// (CDNA5 ISA 10.2: LDS aperture lives in addr[63:32]; addr[31:0] is LDS_ADDR).
__device__ __forceinline__ unsigned lds_addr32(const void* p) {
    return (unsigned)(unsigned long long)p;
}

// ---------------------------------------------------------------------------
// Elementwise f32 -> bf16 convert
// ---------------------------------------------------------------------------
__global__ __launch_bounds__(256)
void cvt_f32_bf16_kernel(const float* __restrict__ in, bf16s* __restrict__ out, size_t n) {
    for (size_t i = (size_t)blockIdx.x * 256 + threadIdx.x; i < n;
         i += (size_t)gridDim.x * 256)
        out[i] = f2bf(in[i]);
}

// ---------------------------------------------------------------------------
// Transpose + convert: in [K, N] f32 -> out [N, K] bf16  (K, N multiples of 32)
// ---------------------------------------------------------------------------
__global__ __launch_bounds__(256)
void transpose_cvt_kernel(const float* __restrict__ in, bf16s* __restrict__ out,
                          int K, int N) {
    __shared__ float t[32][33];
    const int bk = blockIdx.x * 32, bn = blockIdx.y * 32;
    const int tx = threadIdx.x & 31, ty = threadIdx.x >> 5;   // ty in 0..7
#pragma unroll
    for (int i = 0; i < 32; i += 8)
        t[ty + i][tx] = in[(size_t)(bk + ty + i) * N + (bn + tx)];
    __syncthreads();
#pragma unroll
    for (int i = 0; i < 32; i += 8)
        out[(size_t)(bn + ty + i) * K + (bk + tx)] = f2bf(t[tx][ty + i]);
}

// ---------------------------------------------------------------------------
// Codebook row squared norms: cb [1024, 256] f32 -> cb2 [1024]
// ---------------------------------------------------------------------------
__global__ __launch_bounds__(256)
void rownorm_kernel(const float* __restrict__ cb, float* __restrict__ cb2) {
    const int wave = threadIdx.x >> 5, lane = threadIdx.x & 31;
    const int row = blockIdx.x * 8 + wave;
    const float* p = cb + (size_t)row * 256;
    float s = 0.f;
#pragma unroll
    for (int i = 0; i < 8; i++) { float x = p[lane + i * 32]; s += x * x; }
#pragma unroll
    for (int o = 16; o > 0; o >>= 1) s += __shfl_xor(s, o, 32);
    if (lane == 0) cb2[row] = s;
}

// ---------------------------------------------------------------------------
// GEMM: C[M,ND] = A[M,KD](bf16) @ Wt[ND,KD]^T(bf16) + bias, f32 accumulate.
// Block = 256 threads (8 waves), tile 256x64; wave tile 64x32 (8 WMMA accs).
// A tile staged memory->LDS with GLOBAL_LOAD_ASYNC_TO_LDS_B128, double
// buffered: next tile's async copies issued before waiting on the current
// tile (async loads complete in order => s_wait_asynccnt 4 releases the
// current tile while 4 copies for the next tile stay in flight).
// B fragments are read directly from global (weights are tiny / L2-resident).
// ---------------------------------------------------------------------------
template<int KD, int ND, bool WF32, bool WBF16>
__global__ __launch_bounds__(256)
void gemm_bias_kernel(const bf16s* __restrict__ A,
                      const bf16s* __restrict__ Wt,
                      const float* __restrict__ bias,
                      float* __restrict__ Cf,
                      bf16s* __restrict__ Cb) {
    constexpr int APAD = 40;                       // row stride (elems), 16B aligned
    constexpr int NK   = KD / 32;
    __shared__ __align__(16) bf16s sA[2][256 * APAD];

    const int tid  = threadIdx.x;
    const int wave = tid >> 5, lane = tid & 31;
    const int half = lane >> 4, l16 = lane & 15;
    const int wm = wave & 3, wn = wave >> 2;
    const size_t m0 = (size_t)blockIdx.x * 256;
    const int n0 = blockIdx.y * 64;

    // Staging assignment: thread covers rows srow + {0,64,128,192}, chunk sch.
    const int srow = tid >> 2, sch = tid & 3;
    const bf16s* gbase = A + (m0 + srow) * KD + sch * 8;

    auto issue_async = [&](int kt, int buf) {
        const unsigned lbase = lds_addr32(&sA[buf][srow * APAD + sch * 8]);
#pragma unroll
        for (int i = 0; i < 4; i++) {
            const bf16s* gp = gbase + (size_t)i * 64 * KD + kt * 32;
            const unsigned la = lbase + (unsigned)(i * 64 * APAD * 2);
            asm volatile("global_load_async_to_lds_b128 %0, %1, off"
                         :: "v"(la), "v"(gp) : "memory");
        }
    };

    v8f acc[4][2];
#pragma unroll
    for (int i = 0; i < 4; i++)
#pragma unroll
        for (int j = 0; j < 2; j++)
#pragma unroll
            for (int k = 0; k < 8; k++) acc[i][j][k] = 0.f;

    issue_async(0, 0);

#pragma unroll
    for (int t = 0; t < NK; t++) {
        const int cur = t & 1;
        if (t + 1 < NK) {
            issue_async(t + 1, cur ^ 1);
            asm volatile("s_wait_asynccnt 0x4" ::: "memory");
        } else {
            asm volatile("s_wait_asynccnt 0x0" ::: "memory");
        }
        __syncthreads();   // current tile visible to all waves

        const int k0 = t * 32;
        // A fragments (16x32 bf16): half 0 -> K{0..15}, half 1 -> K{16..31}
        v16us afr[4];
#pragma unroll
        for (int mt = 0; mt < 4; mt++) {
            const bf16s* ap = &sA[cur][(wm * 64 + mt * 16 + l16) * APAD + half * 16];
            v8us lo = *(const v8us*)ap;
            v8us hi = *(const v8us*)(ap + 8);
#pragma unroll
            for (int e = 0; e < 8; e++) { afr[mt][e] = lo[e]; afr[mt][8 + e] = hi[e]; }
        }
        // B fragments (32x16 bf16): lane = column N, contiguous K in Wt[N,K]
        v16us bfr[2];
#pragma unroll
        for (int nt = 0; nt < 2; nt++) {
            const bf16s* bp = Wt + (size_t)(n0 + wn * 32 + nt * 16 + l16) * KD
                                 + k0 + half * 16;
            bfr[nt] = *(const v16us*)bp;
        }
#pragma unroll
        for (int mt = 0; mt < 4; mt++)
#pragma unroll
            for (int nt = 0; nt < 2; nt++)
                acc[mt][nt] = wmma_bf16(afr[mt], bfr[nt], acc[mt][nt]);

        __syncthreads();   // all reads done before this buffer is rewritten
    }

    // Epilogue: + bias, store f32 and/or bf16 (C layout: VGPR r -> M = half*8 + r)
#pragma unroll
    for (int nt = 0; nt < 2; nt++) {
        const int n = n0 + wn * 32 + nt * 16 + l16;
        const float bv = bias[n];
#pragma unroll
        for (int mt = 0; mt < 4; mt++) {
#pragma unroll
            for (int r = 0; r < 8; r++) {
                const size_t m = m0 + wm * 64 + mt * 16 + half * 8 + r;
                const float v = acc[mt][nt][r] + bv;
                if (WF32)  Cf[m * ND + n] = v;
                if (WBF16) Cb[m * ND + n] = f2bf(v);
            }
        }
    }
}

// ---------------------------------------------------------------------------
// LayerNorm + exact-erf GELU, one wave per row of length N (N % 32 == 0).
// In f32, out bf16 (feeds next GEMM).
// ---------------------------------------------------------------------------
template<int N>
__global__ __launch_bounds__(256)
void ln_gelu_kernel(const float* __restrict__ in, const float* __restrict__ g,
                    const float* __restrict__ be, bf16s* __restrict__ out, int M) {
    constexpr int NPL = N / 32;
    const int wave = threadIdx.x >> 5, lane = threadIdx.x & 31;
    const int row = blockIdx.x * 8 + wave;
    if (row >= M) return;
    const float* p = in + (size_t)row * N;
    float v[NPL];
    float s = 0.f;
#pragma unroll
    for (int i = 0; i < NPL; i++) { v[i] = p[i * 32 + lane]; s += v[i]; }
#pragma unroll
    for (int o = 16; o > 0; o >>= 1) s += __shfl_xor(s, o, 32);
    const float mean = s * (1.0f / N);
    float q = 0.f;
#pragma unroll
    for (int i = 0; i < NPL; i++) { float d = v[i] - mean; q += d * d; }
#pragma unroll
    for (int o = 16; o > 0; o >>= 1) q += __shfl_xor(q, o, 32);
    const float rstd = rsqrtf(q * (1.0f / N) + 1e-5f);
    bf16s* op = out + (size_t)row * N;
#pragma unroll
    for (int i = 0; i < NPL; i++) {
        const int c = i * 32 + lane;
        const float y = (v[i] - mean) * rstd * g[c] + be[c];
        const float gel = 0.5f * y * (1.0f + erff(y * 0.70710678118654752f));
        op[c] = f2bf(gel);
    }
}

// ---------------------------------------------------------------------------
// VQ argmin: for 16 rows of z_e (bf16), compute dot with all 1024 codes via
// WMMA (K=256), then argmin of cb2[n] - 2*dot with first-index tie-break.
// Block = 256 threads; wave w covers codes [w*128, w*128+128).
// ---------------------------------------------------------------------------
__global__ __launch_bounds__(256)
void vq_argmin_kernel(const bf16s* __restrict__ ze, const bf16s* __restrict__ cb,
                      const float* __restrict__ cb2, int* __restrict__ indices,
                      int M) {
    const int tid = threadIdx.x, wave = tid >> 5, lane = tid & 31;
    const int half = lane >> 4, l16 = lane & 15;
    const size_t m0 = (size_t)blockIdx.x * 16;
    const int nbase = wave * 128;

    v8f acc[8];
#pragma unroll
    for (int t = 0; t < 8; t++)
#pragma unroll
        for (int k = 0; k < 8; k++) acc[t][k] = 0.f;

    for (int k0 = 0; k0 < 256; k0 += 32) {
        // A fragment direct from global (tiny footprint, L2-hot)
        const bf16s* ap = ze + (m0 + l16) * 256 + k0 + half * 16;
        v8us lo = *(const v8us*)ap;
        v8us hi = *(const v8us*)(ap + 8);
        v16us a;
#pragma unroll
        for (int e = 0; e < 8; e++) { a[e] = lo[e]; a[8 + e] = hi[e]; }
#pragma unroll
        for (int t = 0; t < 8; t++) {
            const bf16s* bp = cb + (size_t)(nbase + t * 16 + l16) * 256 + k0 + half * 16;
            v16us b = *(const v16us*)bp;
            acc[t] = wmma_bf16(a, b, acc[t]);
        }
    }

    // Per-lane argmin over its 8 tiles for each of its 8 rows
    float bv[8]; int bn[8];
#pragma unroll
    for (int r = 0; r < 8; r++) { bv[r] = 3.4e38f; bn[r] = 0x7fffffff; }
#pragma unroll
    for (int t = 0; t < 8; t++) {
        const int n = nbase + t * 16 + l16;
        const float c2 = cb2[n];
#pragma unroll
        for (int r = 0; r < 8; r++) {
            const float d = fmaf(-2.0f, acc[t][r], c2);
            if (d < bv[r] || (d == bv[r] && n < bn[r])) { bv[r] = d; bn[r] = n; }
        }
    }
    // Reduce across the 16 lanes of each half (offsets < 16 stay within half)
#pragma unroll
    for (int o = 1; o < 16; o <<= 1) {
#pragma unroll
        for (int r = 0; r < 8; r++) {
            const float ov = __shfl_xor(bv[r], o, 32);
            const int   on = __shfl_xor(bn[r], o, 32);
            if (ov < bv[r] || (ov == bv[r] && on < bn[r])) { bv[r] = ov; bn[r] = on; }
        }
    }
    __shared__ float sval[8][16];
    __shared__ int   sidx[8][16];
    if (l16 == 0) {
#pragma unroll
        for (int r = 0; r < 8; r++) {
            const int m = half * 8 + r;   // lanes 0..15 -> rows 0..7; 16..31 -> 8..15
            sval[wave][m] = bv[r];
            sidx[wave][m] = bn[r];
        }
    }
    __syncthreads();
    if (tid < 16) {
        float best = sval[0][tid]; int bi = sidx[0][tid];
#pragma unroll
        for (int w = 1; w < 8; w++) {
            const float v = sval[w][tid]; const int i2 = sidx[w][tid];
            if (v < best || (v == best && i2 < bi)) { best = v; bi = i2; }
        }
        indices[m0 + tid] = bi;
    }
}

// ---------------------------------------------------------------------------
// Gather z_q + straight-through: zq_st = ze + (zq - ze); also bf16 for decoder
// ---------------------------------------------------------------------------
__global__ __launch_bounds__(256)
void gather_zq_kernel(const int* __restrict__ idx, const float* __restrict__ cb,
                      const float* __restrict__ ze, float* __restrict__ zq_st,
                      bf16s* __restrict__ zqb, size_t n) {
    for (size_t i = (size_t)blockIdx.x * 256 + threadIdx.x; i < n;
         i += (size_t)gridDim.x * 256) {
        const size_t b = i >> 8; const int d = (int)(i & 255);
        const float q = cb[(size_t)idx[b] * 256 + d];
        const float e = ze[i];
        const float st = e + (q - e);
        zq_st[i] = st;
        zqb[i] = f2bf(st);
    }
}

// ---------------------------------------------------------------------------
// Deterministic two-stage MSE reductions
// ---------------------------------------------------------------------------
__global__ __launch_bounds__(256)
void sqdiff_partial_kernel(const float* __restrict__ a, const float* __restrict__ b,
                           size_t n, float* __restrict__ partial) {
    float s = 0.f;
    for (size_t i = (size_t)blockIdx.x * 256 + threadIdx.x; i < n;
         i += (size_t)gridDim.x * 256) {
        const float d = a[i] - b[i];
        s += d * d;
    }
    __shared__ float red[8];
    const int wave = threadIdx.x >> 5, lane = threadIdx.x & 31;
#pragma unroll
    for (int o = 16; o > 0; o >>= 1) s += __shfl_xor(s, o, 32);
    if (lane == 0) red[wave] = s;
    __syncthreads();
    if (threadIdx.x == 0) {
        float t = 0.f;
#pragma unroll
        for (int w = 0; w < 8; w++) t += red[w];
        partial[blockIdx.x] = t;
    }
}

__global__ __launch_bounds__(256)
void finalize_losses_kernel(const float* __restrict__ pA, const float* __restrict__ pB,
                            int nblk, float invA, float invB_comm,
                            float* __restrict__ out3) {
    float sa = 0.f, sb = 0.f;
    for (int i = threadIdx.x; i < nblk; i += 256) { sa += pA[i]; sb += pB[i]; }
    __shared__ float ra[8], rb[8];
    const int wave = threadIdx.x >> 5, lane = threadIdx.x & 31;
#pragma unroll
    for (int o = 16; o > 0; o >>= 1) { sa += __shfl_xor(sa, o, 32); sb += __shfl_xor(sb, o, 32); }
    if (lane == 0) { ra[wave] = sa; rb[wave] = sb; }
    __syncthreads();
    if (threadIdx.x == 0) {
        float ta = 0.f, tb = 0.f;
#pragma unroll
        for (int w = 0; w < 8; w++) { ta += ra[w]; tb += rb[w]; }
        const float recon = ta * invA;
        const float vq    = tb * invB_comm;   // includes 0.25 commitment cost
        out3[0] = recon + vq;
        out3[1] = recon;
        out3[2] = vq;
    }
}

// ---------------------------------------------------------------------------
// Host-side orchestration
// ---------------------------------------------------------------------------
extern "C" void kernel_launch(void* const* d_in, const int* in_sizes, int n_in,
                              void* d_out, int out_size, void* d_ws, size_t ws_size,
                              hipStream_t stream) {
    const float* x      = (const float*)d_in[0];
    const float* eW1    = (const float*)d_in[1];
    const float* eb1    = (const float*)d_in[2];
    const float* eg1    = (const float*)d_in[3];
    const float* ebe1   = (const float*)d_in[4];
    const float* eW2    = (const float*)d_in[5];
    const float* eb2    = (const float*)d_in[6];
    const float* eg2    = (const float*)d_in[7];
    const float* ebe2   = (const float*)d_in[8];
    const float* eW3    = (const float*)d_in[9];
    const float* eb3    = (const float*)d_in[10];
    const float* cbf    = (const float*)d_in[11];
    const float* dW1    = (const float*)d_in[12];
    const float* db1    = (const float*)d_in[13];
    const float* dg1    = (const float*)d_in[14];
    const float* dbe1   = (const float*)d_in[15];
    const float* dW2    = (const float*)d_in[16];
    const float* db2    = (const float*)d_in[17];
    const float* dg2    = (const float*)d_in[18];
    const float* dbe2   = (const float*)d_in[19];
    const float* dW3    = (const float*)d_in[20];
    const float* db3    = (const float*)d_in[21];

    const int B = in_sizes[0] / 768;           // 65536

    // d_out layout: x_recon[B*768], total, recon, vq, indices[B] (int32 bits),
    //               z_e[B*256], z_q_st[B*256]
    float* out    = (float*)d_out;
    float* xrec   = out;
    float* losses = out + (size_t)B * 768;
    int*   idx    = (int*)(losses + 3);
    float* ze_out = losses + 3 + B;
    float* zq_out = ze_out + (size_t)B * 256;

    // Workspace carve-up
    char* w = (char*)d_ws;
    size_t off = 0;
    auto alloc = [&](size_t bytes) -> void* {
        void* p = w + off;
        off = (off + bytes + 255) & ~(size_t)255;
        return p;
    };
    bf16s* xb  = (bf16s*)alloc((size_t)B * 768 * 2);
    bf16s* w1t = (bf16s*)alloc((size_t)512 * 768 * 2);
    bf16s* w2t = (bf16s*)alloc((size_t)384 * 512 * 2);
    bf16s* w3t = (bf16s*)alloc((size_t)256 * 384 * 2);
    bf16s* d1t = (bf16s*)alloc((size_t)384 * 256 * 2);
    bf16s* d2t = (bf16s*)alloc((size_t)512 * 384 * 2);
    bf16s* d3t = (bf16s*)alloc((size_t)768 * 512 * 2);
    bf16s* cbb = (bf16s*)alloc((size_t)1024 * 256 * 2);
    float* cb2 = (float*)alloc((size_t)1024 * 4);
    float* hf  = (float*)alloc((size_t)B * 512 * 4);   // reused f32 GEMM out
    bf16s* h1b = (bf16s*)alloc((size_t)B * 512 * 2);   // 512-wide bf16 acts
    bf16s* h2b = (bf16s*)alloc((size_t)B * 384 * 2);   // 384-wide bf16 acts
    bf16s* zeb = (bf16s*)alloc((size_t)B * 256 * 2);
    bf16s* zqb = (bf16s*)alloc((size_t)B * 256 * 2);
    float* pA  = (float*)alloc((size_t)1024 * 4);
    float* pB  = (float*)alloc((size_t)1024 * 4);
    (void)ws_size; (void)n_in; (void)out_size;

    // --- Pre-pass: converts / transposes / codebook norms ---
    cvt_f32_bf16_kernel<<<4096, 256, 0, stream>>>(x, xb, (size_t)B * 768);
    transpose_cvt_kernel<<<dim3(768 / 32, 512 / 32), 256, 0, stream>>>(eW1, w1t, 768, 512);
    transpose_cvt_kernel<<<dim3(512 / 32, 384 / 32), 256, 0, stream>>>(eW2, w2t, 512, 384);
    transpose_cvt_kernel<<<dim3(384 / 32, 256 / 32), 256, 0, stream>>>(eW3, w3t, 384, 256);
    transpose_cvt_kernel<<<dim3(256 / 32, 384 / 32), 256, 0, stream>>>(dW1, d1t, 256, 384);
    transpose_cvt_kernel<<<dim3(384 / 32, 512 / 32), 256, 0, stream>>>(dW2, d2t, 384, 512);
    transpose_cvt_kernel<<<dim3(512 / 32, 768 / 32), 256, 0, stream>>>(dW3, d3t, 512, 768);
    cvt_f32_bf16_kernel<<<512, 256, 0, stream>>>(cbf, cbb, (size_t)1024 * 256);
    rownorm_kernel<<<128, 256, 0, stream>>>(cbf, cb2);

    const int MB = B / 256;   // GEMM grid.x

    // --- Encoder ---
    gemm_bias_kernel<768, 512, true, false><<<dim3(MB, 512 / 64), 256, 0, stream>>>(
        xb, w1t, eb1, hf, nullptr);
    ln_gelu_kernel<512><<<B / 8, 256, 0, stream>>>(hf, eg1, ebe1, h1b, B);
    gemm_bias_kernel<512, 384, true, false><<<dim3(MB, 384 / 64), 256, 0, stream>>>(
        h1b, w2t, eb2, hf, nullptr);
    ln_gelu_kernel<384><<<B / 8, 256, 0, stream>>>(hf, eg2, ebe2, h2b, B);
    gemm_bias_kernel<384, 256, true, true><<<dim3(MB, 256 / 64), 256, 0, stream>>>(
        h2b, w3t, eb3, ze_out, zeb);

    // --- VQ ---
    vq_argmin_kernel<<<B / 16, 256, 0, stream>>>(zeb, cbb, cb2, idx, B);
    gather_zq_kernel<<<8192, 256, 0, stream>>>(idx, cbf, ze_out, zq_out, zqb,
                                               (size_t)B * 256);

    // --- Decoder ---
    gemm_bias_kernel<256, 384, true, false><<<dim3(MB, 384 / 64), 256, 0, stream>>>(
        zqb, d1t, db1, hf, nullptr);
    ln_gelu_kernel<384><<<B / 8, 256, 0, stream>>>(hf, dg1, dbe1, h2b, B);
    gemm_bias_kernel<384, 512, true, false><<<dim3(MB, 512 / 64), 256, 0, stream>>>(
        h2b, d2t, db2, hf, nullptr);
    ln_gelu_kernel<512><<<B / 8, 256, 0, stream>>>(hf, dg2, dbe2, h1b, B);
    gemm_bias_kernel<512, 768, true, false><<<dim3(MB, 768 / 64), 256, 0, stream>>>(
        h1b, d3t, db3, xrec, nullptr);

    // --- Losses ---
    sqdiff_partial_kernel<<<1024, 256, 0, stream>>>(xrec, x, (size_t)B * 768, pA);
    sqdiff_partial_kernel<<<1024, 256, 0, stream>>>(zq_out, ze_out, (size_t)B * 256, pB);
    finalize_losses_kernel<<<1, 256, 0, stream>>>(
        pA, pB, 1024, 1.0f / ((float)B * 768.0f),
        0.25f / ((float)B * 256.0f), losses);
}
